// LSTM_layer_20177756357054
// MI455X (gfx1250) — matrix operand order repaired and checked by reading, not compiled, not run on hardware
//
#include <hip/hip_runtime.h>

// ---------------- problem geometry (fixed by the reference) ----------------
#define B_        64
#define T_        1024
#define D_        512
#define H_        512
#define NG_       2048            // 4*H
// ---------------- decomposition -------------------------------------------
#define CLUSTERS_ 4               // batch partition: 16 rows per cluster
#define WGPC_     16              // workgroups per cluster (hidden partition)
#define MROWS_    16              // batch rows per cluster
#define HSL_      32              // hidden units per WG
#define NCOLS_    128             // gate columns per WG = 4*HSL_  [i|f|g|o]
#define KTOT_     1024            // D_ + H_  (A = [x_t | h])
#define THREADS_  256             // 8 waves (wave32)
// LDS: pad bf16 row stride (+8 elems = +16B) to break 2KB-stride bank conflicts
#define WSTRIDE_  (KTOT_ + 8)
#define ASTRIDE_  (KTOT_ + 8)
#define WLDS_BYTES (NCOLS_ * WSTRIDE_ * 2)      // 264192
#define ABUF_BYTES (MROWS_ * ASTRIDE_ * 2)      //  33024
#define GBUF_BYTES (MROWS_ * NCOLS_ * 4)        //   8192
#define CBUF_BYTES (MROWS_ * HSL_ * 4)          //   2048
#define BIAS_BYTES (NCOLS_ * 4)                 //    512
#define SMEM_BYTES (WLDS_BYTES + ABUF_BYTES + GBUF_BYTES + CBUF_BYTES + BIAS_BYTES)

typedef __attribute__((ext_vector_type(16))) __bf16 v16bf;
typedef __attribute__((ext_vector_type(8)))  __bf16 v8bf;
typedef __attribute__((ext_vector_type(4)))  __bf16 v4bf;
typedef __attribute__((ext_vector_type(8)))  float  v8f;
typedef __attribute__((ext_vector_type(4)))  float  v4f;

__device__ __forceinline__ float fast_sigmoid(float x) {
  return 1.0f / (1.0f + __expf(-x));
}
__device__ __forceinline__ float fast_tanh(float x) {
  return 1.0f - 2.0f / (__expf(2.0f * x) + 1.0f);
}

__global__ void zero_ws(unsigned int* p, int n) {
  int i = blockIdx.x * blockDim.x + threadIdx.x;
  if (i < n) p[i] = 0u;
}

// Persistent LSTM: 64 WGs = 4 clusters x 16 WGs. Weights live in LDS (bf16,
// column-major), h is exchanged per step through a double-buffered L2 region
// with a per-cluster barrier (atomic counter + s_cluster_barrier fast path).
__global__ __launch_bounds__(THREADS_, 1)
void lstm_persistent(const float* __restrict__ x,  const float* __restrict__ h0,
                     const float* __restrict__ c0, const float* __restrict__ Wx,
                     const float* __restrict__ Wh, const float* __restrict__ bias_g,
                     float* __restrict__ out, unsigned int* __restrict__ bar,
                     __bf16* __restrict__ hxch /* [2][CLUSTERS_][MROWS_][H_] */) {
  extern __shared__ char smem[];
  __bf16* wlds = reinterpret_cast<__bf16*>(smem);                          // [NCOLS_][WSTRIDE_]
  __bf16* abuf = reinterpret_cast<__bf16*>(smem + WLDS_BYTES);             // [MROWS_][ASTRIDE_]
  float*  gbuf = reinterpret_cast<float*>(smem + WLDS_BYTES + ABUF_BYTES); // [MROWS_][NCOLS_]
  float*  cbuf = gbuf + MROWS_ * NCOLS_;                                   // [MROWS_][HSL_]
  float*  blds = cbuf + MROWS_ * HSL_;                                     // [NCOLS_]

  const int tid  = threadIdx.x;
  const int wg   = blockIdx.x;
  const int cl   = wg >> 4;                 // cluster id 0..3  (batch slice)
  const int wid  = wg & 15;                 // wg-in-cluster    (hidden slice)
  const int row0 = cl * MROWS_;
  const int col0 = wid * HSL_;

  // ---- prologue: stage [Wx; Wh] slice into LDS, bf16, column-major ----
  for (int idx = tid; idx < NCOLS_ * KTOT_; idx += THREADS_) {
    int k = idx >> 7;                       // 0..1023
    int n = idx & (NCOLS_ - 1);             // local gate column
    int gc = ((n >> 5) * H_) + col0 + (n & 31);   // global gate column
    float v = (k < D_) ? Wx[(size_t)k * NG_ + gc]
                       : Wh[(size_t)(k - D_) * NG_ + gc];
    wlds[n * WSTRIDE_ + k] = (__bf16)v;
  }
  if (tid < NCOLS_) {
    int gc = ((tid >> 5) * H_) + col0 + (tid & 31);
    blds[tid] = bias_g[gc];
  }
  // h0 -> abuf[:, D_:]  (vectorized: 4 floats -> 4 bf16 per op)
  for (int idx = tid; idx < MROWS_ * H_ / 4; idx += THREADS_) {
    int r = idx >> 7, k4 = (idx & 127) * 4;
    v4f f = *reinterpret_cast<const v4f*>(h0 + (size_t)(row0 + r) * H_ + k4);
    v4bf p = { (__bf16)f.x, (__bf16)f.y, (__bf16)f.z, (__bf16)f.w };
    *reinterpret_cast<v4bf*>(abuf + r * ASTRIDE_ + D_ + k4) = p;
  }
  // x(t=0) -> abuf[:, :D_]
  for (int idx = tid; idx < MROWS_ * D_ / 4; idx += THREADS_) {
    int r = idx >> 7, k4 = (idx & 127) * 4;
    v4f f = *reinterpret_cast<const v4f*>(x + ((size_t)(row0 + r) * T_ + 0) * D_ + k4);
    v4bf p = { (__bf16)f.x, (__bf16)f.y, (__bf16)f.z, (__bf16)f.w };
    *reinterpret_cast<v4bf*>(abuf + r * ASTRIDE_ + k4) = p;
  }
  for (int idx = tid; idx < MROWS_ * HSL_; idx += THREADS_) {
    int r = idx >> 5, j = idx & (HSL_ - 1);
    cbuf[idx] = c0[(size_t)(row0 + r) * H_ + col0 + j];
  }
  __syncthreads();

  // per-lane WMMA addressing (16-bit A 16x32 / B 32x16 layouts, wave32)
  const int lane = tid & 31;
  const int wave = tid >> 5;
  const int n0   = wave * 16;               // this wave's 16 gate columns
  const int arow = lane & 15;               // M (A) / N (B)
  const int hi   = lane >> 4;               // lane group 0/1
  const __bf16* aptr = abuf + arow * ASTRIDE_ + hi * 16;
  const __bf16* bptr = wlds + (n0 + arow) * WSTRIDE_ + hi * 16;

  auto ldA = [&](int kt) -> v16bf {
    const __bf16* ap = aptr + kt * 32;
    v8bf lo  = *reinterpret_cast<const v8bf*>(ap);        // K kb..kb+7
    v8bf hiv = *reinterpret_cast<const v8bf*>(ap + 8);    // K kb+8..kb+15
    return __builtin_shufflevector(lo, hiv,
        0, 1, 2, 3, 4, 5, 6, 7, 8, 9, 10, 11, 12, 13, 14, 15);
  };
  auto ldB = [&](int kt) -> v16bf {
    return *reinterpret_cast<const v16bf*>(bptr + kt * 32);
  };

  for (int t = 0; t < T_; ++t) {
    // ---- gates[16,128] = [x_t | h] (16x1024) @ Wslice (1024x128) ----
    // Two independent accumulator chains (break WMMA->WMMA RAW through C)
    // plus an explicit 2-deep register pipeline so ds_load_b128 for k-tiles
    // kt+2/kt+3 are in flight while the matrix pipe chews on kt/kt+1.
    v8f acc0 = {0.f, 0.f, 0.f, 0.f, 0.f, 0.f, 0.f, 0.f};
    v8f acc1 = {0.f, 0.f, 0.f, 0.f, 0.f, 0.f, 0.f, 0.f};
    v16bf a0 = ldA(0), b0 = ldB(0);
    v16bf a1 = ldA(1), b1 = ldB(1);
#pragma unroll 5
    for (int kt = 0; kt < KTOT_ / 32 - 2; kt += 2) {      // 15 steady iterations
      v16bf a2 = ldA(kt + 2), b2 = ldB(kt + 2);
      v16bf a3 = ldA(kt + 3), b3 = ldB(kt + 3);
      acc0 = __builtin_amdgcn_wmma_f32_16x16x32_bf16(false, a0, false, b0,
                                                     (short)0, acc0, false, false);
      acc1 = __builtin_amdgcn_wmma_f32_16x16x32_bf16(false, a1, false, b1,
                                                     (short)0, acc1, false, false);
      a0 = a2; b0 = b2; a1 = a3; b1 = b3;
    }
    acc0 = __builtin_amdgcn_wmma_f32_16x16x32_bf16(false, a0, false, b0,
                                                   (short)0, acc0, false, false);
    acc1 = __builtin_amdgcn_wmma_f32_16x16x32_bf16(false, a1, false, b1,
                                                   (short)0, acc1, false, false);
#pragma unroll
    for (int v = 0; v < 8; ++v) {           // C/D layout: VGPR v -> M = v + 8*hi
      gbuf[(v + hi * 8) * NCOLS_ + n0 + arow] = acc0[v] + acc1[v];
    }
    __syncthreads();

    // ---- pointwise LSTM cell (2 cells per thread of the 16x32 tile) ----
    for (int idx = tid; idx < MROWS_ * HSL_; idx += THREADS_) {
      int r = idx >> 5, j = idx & 31;
      const float* g = gbuf + r * NCOLS_;
      float si = fast_sigmoid(g[j]       + blds[j]);
      float sf = fast_sigmoid(g[32 + j]  + blds[32 + j]);
      float tg = fast_tanh  (g[64 + j]   + blds[64 + j]);
      float so = fast_sigmoid(g[96 + j]  + blds[96 + j]);
      float cn = sf * cbuf[idx] + si * tg;
      cbuf[idx] = cn;
      float hn = so * fast_tanh(cn);
      __builtin_nontemporal_store(hn,
          out + ((size_t)(row0 + r) * T_ + t) * H_ + col0 + j);
      hxch[(((size_t)(t & 1) * CLUSTERS_ + cl) * MROWS_ + r) * H_ + col0 + j] = (__bf16)hn;
    }
    // warm L2 for x_{t+2} (global_prefetch_b8)
    if (t + 2 < T_ && tid < 64) {
      int r = tid >> 2, seg = tid & 3;
      __builtin_prefetch(x + ((size_t)(row0 + r) * T_ + (t + 2)) * D_ + seg * 128, 0, 0);
    }

    // ---- cluster-wide hand-off of h ----
    __builtin_amdgcn_fence(__ATOMIC_RELEASE, "agent");
    if (tid == 0) {
      unsigned int* c = bar + (size_t)cl * T_ + t;   // one-shot counter per step
      __hip_atomic_fetch_add(c, 1u, __ATOMIC_RELEASE, __HIP_MEMORY_SCOPE_AGENT);
      while (__hip_atomic_load(c, __ATOMIC_ACQUIRE, __HIP_MEMORY_SCOPE_AGENT) < WGPC_)
        __builtin_amdgcn_s_sleep(2);
    }
    if (wave == 0) {
      // hardware fast path when dispatched as a 16-WG cluster (NOP otherwise)
#if defined(__has_builtin) && __has_builtin(__builtin_amdgcn_s_cluster_barrier)
      __builtin_amdgcn_s_cluster_barrier();
#else
      asm volatile("s_barrier_signal -3\n\ts_barrier_wait -3" ::: "memory");
#endif
    }
    __syncthreads();
    __builtin_amdgcn_fence(__ATOMIC_ACQUIRE, "agent");

    // ---- stage operands for t+1 (vectorized b128 copies: critical path) ----
    if (t + 1 < T_) {
      const __bf16* src = hxch + ((size_t)(t & 1) * CLUSTERS_ + cl) * MROWS_ * H_;
      for (int idx = tid; idx < MROWS_ * H_ / 8; idx += THREADS_) {
        int r = idx >> 6, k8 = (idx & 63) * 8;
        *reinterpret_cast<v8bf*>(abuf + r * ASTRIDE_ + D_ + k8) =
            *reinterpret_cast<const v8bf*>(src + r * H_ + k8);
      }
      for (int idx = tid; idx < MROWS_ * D_ / 4; idx += THREADS_) {
        int r = idx >> 7, k4 = (idx & 127) * 4;
        v4f f = *reinterpret_cast<const v4f*>(
            x + ((size_t)(row0 + r) * T_ + (t + 1)) * D_ + k4);
        v4bf p = { (__bf16)f.x, (__bf16)f.y, (__bf16)f.z, (__bf16)f.w };
        *reinterpret_cast<v4bf*>(abuf + r * ASTRIDE_ + k4) = p;
      }
      __syncthreads();
    }
  }
}

extern "C" void kernel_launch(void* const* d_in, const int* in_sizes, int n_in,
                              void* d_out, int out_size, void* d_ws, size_t ws_size,
                              hipStream_t stream) {
  (void)in_sizes; (void)n_in; (void)out_size; (void)ws_size;
  const float* x  = (const float*)d_in[0];
  const float* h0 = (const float*)d_in[1];
  const float* c0 = (const float*)d_in[2];
  const float* Wx = (const float*)d_in[3];
  const float* Wh = (const float*)d_in[4];
  const float* b  = (const float*)d_in[5];
  float* out = (float*)d_out;

  // workspace: [0, 16KB) per-step barrier counters; then double-buffered h exchange
  unsigned int* bar = (unsigned int*)d_ws;
  __bf16* hxch = (__bf16*)((char*)d_ws + (size_t)CLUSTERS_ * T_ * sizeof(unsigned int));

  zero_ws<<<dim3((CLUSTERS_ * T_ + 255) / 256), dim3(256), 0, stream>>>(
      bar, CLUSTERS_ * T_);

  static_assert(SMEM_BYTES <= 320 * 1024, "exceeds 320KB WGP LDS");
  (void)hipFuncSetAttribute(reinterpret_cast<const void*>(lstm_persistent),
                            hipFuncAttributeMaxDynamicSharedMemorySize, SMEM_BYTES);
  lstm_persistent<<<dim3(CLUSTERS_ * WGPC_), dim3(THREADS_), SMEM_BYTES, stream>>>(
      x, h0, c0, Wx, Wh, b, out, bar, hxch);
}
